// LinearTrmAttention_45595372814568
// MI455X (gfx1250) — matrix operand order repaired and checked by reading, not compiled, not run on hardware
//
#include <hip/hip_runtime.h>
#include <cstdint>

// ---------------------------------------------------------------------------
// LinearTrmAttention for MI455X (gfx1250, wave32, WMMA bf16 + TDM async tiles)
// B=8, S=4096, H=1024, NH=16, HD=64
// ---------------------------------------------------------------------------

#define Bq 8
#define Sq 4096
#define Hq 1024
#define NHq 16
#define HDq 64
#define Mrows (Bq * Sq)          // 32768

typedef __bf16 bf16_t;
typedef __attribute__((ext_vector_type(16))) __bf16 v16bf;
typedef __attribute__((ext_vector_type(8)))  __bf16 v8bf;
typedef __attribute__((ext_vector_type(4)))  __bf16 v4bf;
typedef __attribute__((ext_vector_type(8)))  float  v8f;
typedef unsigned int u32x4 __attribute__((ext_vector_type(4)));
typedef int          i32x4 __attribute__((ext_vector_type(4)));
typedef int          i32x8 __attribute__((ext_vector_type(8)));

#ifndef __has_builtin
#define __has_builtin(x) 0
#endif

// Tensor Data Mover availability (device pass for gfx1250 only; host pass and
// other targets take the manual staging fallback).
#if defined(__gfx1250__) && __has_builtin(__builtin_amdgcn_tensor_load_to_lds) && \
    __has_builtin(__builtin_amdgcn_s_wait_tensorcnt)
#define LTA_TDM 1
#if __has_include(<hip/amd_detail/amd_gfx1250_TDM.h>)
#define LTA_TDM_6ARG 1   // amdgpu-toolchain (therock headers): 6-arg builtin
#endif
#else
#define LTA_TDM 0
#endif

static __device__ __forceinline__ bf16_t f2bf(float f) {
  union { float f; uint32_t u; } a; a.f = f;
  uint32_t r = a.u + 0x7FFFu + ((a.u >> 16) & 1u);   // RNE
  unsigned short h = (unsigned short)(r >> 16);
  return __builtin_bit_cast(bf16_t, h);
}
static __device__ __forceinline__ float bf2f(bf16_t b) {
  unsigned short h = __builtin_bit_cast(unsigned short, b);
  union { uint32_t u; float f; } a; a.u = ((uint32_t)h) << 16;
  return a.f;
}
static __device__ __forceinline__ v8bf ld8(const bf16_t* p) {
  return *(const v8bf*)p;
}
static __device__ __forceinline__ v16bf mk16(v8bf lo, v8bf hi) {
  v16bf r;
#pragma unroll
  for (int i = 0; i < 8; ++i) { r[i] = lo[i]; r[8 + i] = hi[i]; }
  return r;
}
static __device__ __forceinline__ v8f wmma_bf16(v16bf a, v16bf b, v8f c) {
  return __builtin_amdgcn_wmma_f32_16x16x32_bf16(
      /*neg_a=*/false, a, /*neg_b=*/false, b,
      /*c_mod=*/(short)0, c, /*reuse_a=*/false, /*reuse_b=*/false);
}

#if LTA_TDM
// Low 32 bits of a generic pointer to LDS == LDS byte offset (ISA aperture map)
static __device__ __forceinline__ uint32_t lds_off_of(const void* p) {
  return (uint32_t)(uintptr_t)p;
}
static __device__ __forceinline__ void lta_tdm_call(u32x4 g0, i32x8 g1) {
  i32x4 z4 = {0, 0, 0, 0};
#ifdef LTA_TDM_6ARG
  i32x8 z8 = {0, 0, 0, 0, 0, 0, 0, 0};
  __builtin_amdgcn_tensor_load_to_lds(g0, g1, z4, z4, z8, 0);
#else
  __builtin_amdgcn_tensor_load_to_lds(g0, g1, z4, z4, 0);
#endif
}
// 2-D tile DMA: tile_d1 lines of tile_d0_u 8-byte units; line stride stride_u
// units; LDS row padding via pad_interval/pad_amount (DWORD encodings per ISA).
static __device__ __forceinline__ void lta_tdm_load_2d(
    const void* gsrc, uint32_t lds_off,
    uint32_t tile_d0_u, uint32_t tile_d1, uint32_t stride_u,
    uint32_t tdim0_u, uint32_t tdim1,
    uint32_t pad_interval, uint32_t pad_amount) {
  uint64_t ga = (uint64_t)(uintptr_t)gsrc;
  u32x4 g0;
  g0[0] = 1u;  // count=1, user descriptor, no gather
  g0[1] = (uint32_t)__builtin_amdgcn_readfirstlane((int)lds_off);
  g0[2] = (uint32_t)__builtin_amdgcn_readfirstlane((int)(uint32_t)ga);
  g0[3] = (uint32_t)__builtin_amdgcn_readfirstlane(
              (int)((uint32_t)(ga >> 32) | (2u << 30)));  // addr[56:32] | type=2
  i32x8 g1;
  // d0: data_size=3 (8B) | pad_enable | pad_interval | pad_amount
  g1[0] = (int)((3u << 16) | (1u << 20) | (pad_interval << 22) | (pad_amount << 25));
  g1[1] = (int)((tdim0_u & 0xFFFFu) << 16);                         // tensor_dim0 lo16
  g1[2] = (int)((tdim0_u >> 16) | ((tdim1 & 0xFFFFu) << 16));       // td0 hi16 | td1 lo16
  g1[3] = (int)((tdim1 >> 16) | ((tile_d0_u & 0xFFFFu) << 16));     // td1 hi16 | tile_dim0
  g1[4] = (int)(tile_d1 & 0xFFFFu);                                 // tile_dim1 (tile_dim2=0)
  g1[5] = (int)stride_u;                                            // tensor_dim0_stride lo32
  g1[6] = 0;
  g1[7] = 0;
  lta_tdm_call(g0, g1);
}
#endif  // LTA_TDM

// ---------------------------------------------------------------------------
// 1) fp32 -> bf16 conversion (vectorized)
// ---------------------------------------------------------------------------
__global__ __launch_bounds__(256) void lta_cvt_bf16(const float* __restrict__ src,
                                                    bf16_t* __restrict__ dst, int n) {
  int i = (blockIdx.x * 256 + threadIdx.x) * 4;
  if (i + 3 < n) {
    float4 v = *(const float4*)(src + i);
    v4bf o; o[0] = f2bf(v.x); o[1] = f2bf(v.y); o[2] = f2bf(v.z); o[3] = f2bf(v.w);
    *(v4bf*)(dst + i) = o;
  }
}

// ---------------------------------------------------------------------------
// 2) Projection GEMM: out = act(A @ W^T + bias)
//    block tile 128x128, 8 waves (2x4); A tile via TDM (double-buffered),
//    W fragments straight from global (2MB bf16, L2-resident).
// ---------------------------------------------------------------------------
__global__ __launch_bounds__(256) void lta_gemm_proj(const bf16_t* __restrict__ A,
                                                     const bf16_t* __restrict__ W,
                                                     const float*  __restrict__ bias,
                                                     bf16_t* __restrict__ out, int act) {
  __shared__ bf16_t aT[2][128][40];   // 80B row stride (TDM pad 16B / 64B)

  const int  tid   = threadIdx.x;
  const int  lane  = tid & 31;
  const int  w     = tid >> 5;
  const int  wm    = w & 1;
  const int  wn    = w >> 1;
  const long rowBlk = (long)blockIdx.x * 128;
  const int  colBlk = blockIdx.y * 128;

  const int lrow  = lane & 15;
  const int lhalf = lane >> 4;
  const int kpA   = lhalf * 8;
  const int kpB   = lhalf * 16;

  v8f c[4][2] = {};

#if LTA_TDM
  const bf16_t* abase = A + rowBlk * Hq;
  if (w == 0)
    lta_tdm_load_2d(abase, lds_off_of(&aT[0][0][0]),
                    /*tile 64B=*/8, /*lines*/128, /*stride 2048B=*/256,
                    /*tdim0*/256, /*tdim1*/(uint32_t)Mrows,
                    /*pad: every 16DW*/3, /*pad 4DW*/3);
  int cur = 0;
#endif

  for (int k0 = 0; k0 < Hq; k0 += 32) {
#if LTA_TDM
    if (w == 0) __builtin_amdgcn_s_wait_tensorcnt(0);
    __syncthreads();
    if (w == 0 && (k0 + 32) < Hq)
      lta_tdm_load_2d(abase + (k0 + 32), lds_off_of(&aT[cur ^ 1][0][0]),
                      8, 128, 256, 256, (uint32_t)Mrows, 3, 3);
#else
    const int cur = 0;
    {
      const int ldRow = tid >> 1;
      const int ldCol = (tid & 1) * 16;
      const bf16_t* gp = A + (rowBlk + ldRow) * Hq + k0 + ldCol;
      v8bf a0 = ld8(gp), a1 = ld8(gp + 8);
      __builtin_prefetch(gp + 32, 0, 3);
      *(v8bf*)&aT[0][ldRow][ldCol]     = a0;
      *(v8bf*)&aT[0][ldRow][ldCol + 8] = a1;
    }
    __syncthreads();
#endif

    v16bf bf[2];
#pragma unroll
    for (int nt = 0; nt < 2; ++nt) {
      const int n = colBlk + wn * 32 + nt * 16 + lrow;
      const bf16_t* wp = W + (long)n * Hq + k0 + kpB;
      bf[nt] = mk16(ld8(wp), ld8(wp + 8));
    }
#pragma unroll
    for (int mt = 0; mt < 4; ++mt) {
      const int r = wm * 64 + mt * 16 + lrow;
      v16bf af = mk16(*(const v8bf*)&aT[cur][r][kpB],
                      *(const v8bf*)&aT[cur][r][kpB + 8]);
#pragma unroll
      for (int nt = 0; nt < 2; ++nt)
        c[mt][nt] = wmma_bf16(af, bf[nt], c[mt][nt]);
    }
    __syncthreads();
#if LTA_TDM
    cur ^= 1;
#endif
  }

#pragma unroll
  for (int mt = 0; mt < 4; ++mt) {
#pragma unroll
    for (int nt = 0; nt < 2; ++nt) {
      const int n  = colBlk + wn * 32 + nt * 16 + lrow;
      const float bn = bias[n];
#pragma unroll
      for (int r = 0; r < 8; ++r) {
        const long row = rowBlk + wm * 64 + mt * 16 + r + 8 * lhalf;
        float v = c[mt][nt][r] + bn;
        if (act) v = (v > 0.f) ? (v + 1.f) : __expf(v);
        out[row * Hq + n] = f2bf(v);
      }
    }
  }
}

// ---------------------------------------------------------------------------
// 3) ksum[b,h,d] = sum_s k[b,s,h,d]
// ---------------------------------------------------------------------------
__global__ __launch_bounds__(256) void lta_ksum(const bf16_t* __restrict__ kb,
                                                float* __restrict__ ks) {
  __shared__ float red[4][64];
  const int bh = blockIdx.x;
  const int b = bh >> 4, h = bh & 15;
  const int d  = threadIdx.x & 63;
  const int sl = threadIdx.x >> 6;
  float acc = 0.f;
  for (int s = sl * 1024; s < sl * 1024 + 1024; ++s)
    acc += bf2f(kb[((long)(b * Sq + s) * NHq + h) * HDq + d]);
  red[sl][d] = acc;
  __syncthreads();
  if (threadIdx.x < 64)
    ks[bh * 64 + threadIdx.x] =
        red[0][threadIdx.x] + red[1][threadIdx.x] + red[2][threadIdx.x] + red[3][threadIdx.x];
}

// ---------------------------------------------------------------------------
// 4) z[b,s,h] = 1 / dot(q[b,s,h,:], ksum[b,h,:] + eps)
// ---------------------------------------------------------------------------
__global__ __launch_bounds__(256) void lta_z(const bf16_t* __restrict__ qb,
                                             const float* __restrict__ ks,
                                             float* __restrict__ z) {
  const long i = (long)blockIdx.x * 256 + threadIdx.x;
  const int  h = (int)(i & 15);
  const long bs = i >> 4;
  const int  b  = (int)(bs >> 12);
  const bf16_t* q = qb + (bs * NHq + h) * HDq;
  const float*  k = ks + (b * NHq + h) * HDq;
  float acc = 0.f;
#pragma unroll 8
  for (int d = 0; d < HDq; ++d) acc += bf2f(q[d]) * (k[d] + 1e-6f);
  z[i] = 1.0f / acc;
}

// ---------------------------------------------------------------------------
// 5) kv[b,h,m,d] = sum_s v[b,s,h,m] * k[b,s,h,d]   (64x64, K=4096)
//    one block (4 waves) per (b,h); operands transposed through LDS
// ---------------------------------------------------------------------------
__global__ __launch_bounds__(128) void lta_kv(const bf16_t* __restrict__ kb,
                                              const bf16_t* __restrict__ vb,
                                              bf16_t* __restrict__ kv) {
  __shared__ bf16_t Kt[64][40];
  __shared__ bf16_t Vt[64][40];
  const int bh = blockIdx.x, b = bh >> 4, h = bh & 15;
  const int tid = threadIdx.x, lane = tid & 31, w = tid >> 5;
  const int lrow = lane & 15, lhalf = lane >> 4;

  v8f c[4] = {};
  const int si = tid & 31;
  const int db = (tid >> 5) * 16;

  for (int s0 = 0; s0 < Sq; s0 += 32) {
    const bf16_t* kp = kb + ((long)(b * Sq + s0 + si) * NHq + h) * HDq + db;
    const bf16_t* vp = vb + ((long)(b * Sq + s0 + si) * NHq + h) * HDq + db;
#pragma unroll
    for (int j = 0; j < 16; ++j) {
      Kt[db + j][si] = kp[j];
      Vt[db + j][si] = vp[j];
    }
    __syncthreads();
    {
      const int r = w * 16 + lrow;
      v16bf af = mk16(*(const v8bf*)&Vt[r][lhalf * 16],
                      *(const v8bf*)&Vt[r][lhalf * 16 + 8]);
#pragma unroll
      for (int nt = 0; nt < 4; ++nt) {
        const int d = nt * 16 + lrow;
        v16bf bfr = mk16(*(const v8bf*)&Kt[d][lhalf * 16],
                         *(const v8bf*)&Kt[d][lhalf * 16 + 8]);
        c[nt] = wmma_bf16(af, bfr, c[nt]);
      }
    }
    __syncthreads();
  }
#pragma unroll
  for (int nt = 0; nt < 4; ++nt)
#pragma unroll
    for (int r = 0; r < 8; ++r) {
      const int m = w * 16 + r + 8 * lhalf;
      kv[((long)bh * 64 + m) * 64 + nt * 16 + lrow] = f2bf(c[nt][r]);
    }
}

// ---------------------------------------------------------------------------
// 6) ctx[b,s,h,m] = z[b,s,h] * sum_d q[b,s,h,d] * kv[b,h,m,d]
//    kv tile (8KB) DMA'd into LDS once via TDM
// ---------------------------------------------------------------------------
__global__ __launch_bounds__(256) void lta_ctx(const bf16_t* __restrict__ qb,
                                               const bf16_t* __restrict__ kv,
                                               const float* __restrict__ z,
                                               bf16_t* __restrict__ ctx) {
  __shared__ bf16_t kvT[64][72];  // 144B row stride (TDM pad 16B / 128B)
  const int bh = blockIdx.y, b = bh >> 4, h = bh & 15;
  const int sBase = blockIdx.x * 128;
  const int tid = threadIdx.x, lane = tid & 31, w = tid >> 5;
  const int lrow = lane & 15, lhalf = lane >> 4;

#if LTA_TDM
  if (w == 0) {
    lta_tdm_load_2d(kv + (long)bh * 64 * 64, lds_off_of(&kvT[0][0]),
                    /*tile 128B=*/16, /*lines*/64, /*stride*/16,
                    /*tdim0*/16, /*tdim1*/64,
                    /*pad: every 32DW*/4, /*pad 4DW*/3);
    __builtin_amdgcn_s_wait_tensorcnt(0);
  }
  __syncthreads();
#else
  {
    const int m = tid >> 2, dB = (tid & 3) * 16;
    const bf16_t* p = kv + ((long)bh * 64 + m) * 64 + dB;
    *(v8bf*)&kvT[m][dB]     = ld8(p);
    *(v8bf*)&kvT[m][dB + 8] = ld8(p + 8);
  }
  __syncthreads();
#endif

  v8f c[4] = {};
  const int srow = sBase + w * 16 + lrow;
  const bf16_t* qrow = qb + ((long)(b * Sq + srow) * NHq + h) * HDq;
#pragma unroll
  for (int kd = 0; kd < 64; kd += 32) {
    v16bf af = mk16(ld8(qrow + kd + lhalf * 16), ld8(qrow + kd + lhalf * 16 + 8));
#pragma unroll
    for (int nt = 0; nt < 4; ++nt) {
      const int m = nt * 16 + lrow;
      v16bf bfr = mk16(*(const v8bf*)&kvT[m][kd + lhalf * 16],
                       *(const v8bf*)&kvT[m][kd + lhalf * 16 + 8]);
      c[nt] = wmma_bf16(af, bfr, c[nt]);
    }
  }
#pragma unroll
  for (int r = 0; r < 8; ++r) {
    const long s  = sBase + w * 16 + r + 8 * lhalf;
    const float zz = z[(long)(b * Sq + s) * NHq + h];
#pragma unroll
    for (int nt = 0; nt < 4; ++nt)
      ctx[((long)(b * Sq + s) * NHq + h) * HDq + nt * 16 + lrow] = f2bf(c[nt][r] * zz);
  }
}

// ---------------------------------------------------------------------------
// 7) out = LayerNorm(ctx @ Wd^T + bd + x) * gamma + beta  (fp32 output)
//    block = 16 rows x full N=1024; A tiles via TDM (double-buffered)
// ---------------------------------------------------------------------------
__global__ __launch_bounds__(256) void lta_out(const bf16_t* __restrict__ ctx,
                                               const bf16_t* __restrict__ Wd,
                                               const float* __restrict__ bd,
                                               const float* __restrict__ xin,
                                               const float* __restrict__ gamma,
                                               const float* __restrict__ beta,
                                               float* __restrict__ out) {
  __shared__ bf16_t aT[2][16][40];
  __shared__ float rsum[16], rsq[16];

  const long rowBase = (long)blockIdx.x * 16;
  const int tid = threadIdx.x, lane = tid & 31, w = tid >> 5;
  const int lrow = lane & 15, lhalf = lane >> 4;

  v8f c[8] = {};

#if LTA_TDM
  const bf16_t* abase = ctx + rowBase * Hq;
  if (w == 0)
    lta_tdm_load_2d(abase, lds_off_of(&aT[0][0][0]),
                    8, 16, 256, 256, 16, 3, 3);
  int cur = 0;
#endif

  for (int k0 = 0; k0 < Hq; k0 += 32) {
#if LTA_TDM
    if (w == 0) __builtin_amdgcn_s_wait_tensorcnt(0);
    __syncthreads();
    if (w == 0 && (k0 + 32) < Hq)
      lta_tdm_load_2d(abase + (k0 + 32), lds_off_of(&aT[cur ^ 1][0][0]),
                      8, 16, 256, 256, 16, 3, 3);
#else
    const int cur = 0;
    {
      const int r = tid >> 4, cp = (tid & 15) * 2;
      const bf16_t* p = ctx + (rowBase + r) * Hq + k0 + cp;
      aT[0][r][cp] = p[0]; aT[0][r][cp + 1] = p[1];
    }
    __syncthreads();
#endif
    v16bf af = mk16(*(const v8bf*)&aT[cur][lrow][lhalf * 16],
                    *(const v8bf*)&aT[cur][lrow][lhalf * 16 + 8]);
#pragma unroll
    for (int nt = 0; nt < 8; ++nt) {
      const int n = w * 128 + nt * 16 + lrow;
      const bf16_t* wp = Wd + (long)n * Hq + k0 + lhalf * 16;
      c[nt] = wmma_bf16(af, mk16(ld8(wp), ld8(wp + 8)), c[nt]);
    }
    __syncthreads();
#if LTA_TDM
    cur ^= 1;
#endif
  }

  float vals[8][8];
#pragma unroll
  for (int nt = 0; nt < 8; ++nt) {
    const int n = w * 128 + nt * 16 + lrow;
    const float bn = bd[n];
#pragma unroll
    for (int r = 0; r < 8; ++r) {
      const long row = rowBase + r + 8 * lhalf;
      vals[nt][r] = c[nt][r] + bn + xin[row * Hq + n];
    }
  }

  if (tid < 16) { rsum[tid] = 0.f; rsq[tid] = 0.f; }
  __syncthreads();
#pragma unroll
  for (int r = 0; r < 8; ++r) {
    float ps = 0.f, pq = 0.f;
#pragma unroll
    for (int nt = 0; nt < 8; ++nt) { ps += vals[nt][r]; pq += vals[nt][r] * vals[nt][r]; }
    const int rowL = r + 8 * lhalf;
    atomicAdd(&rsum[rowL], ps);
    atomicAdd(&rsq[rowL], pq);
  }
  __syncthreads();
#pragma unroll
  for (int r = 0; r < 8; ++r) {
    const int rowL = r + 8 * lhalf;
    const float mu   = rsum[rowL] * (1.0f / 1024.0f);
    const float var  = rsq[rowL] * (1.0f / 1024.0f) - mu * mu;
    const float rstd = rsqrtf(var + 1e-12f);
    const long row = rowBase + rowL;
#pragma unroll
    for (int nt = 0; nt < 8; ++nt) {
      const int n = w * 128 + nt * 16 + lrow;
      out[row * Hq + n] = (vals[nt][r] - mu) * rstd * gamma[n] + beta[n];
    }
  }
}

// ---------------------------------------------------------------------------
// host launcher
// ---------------------------------------------------------------------------
extern "C" void kernel_launch(void* const* d_in, const int* in_sizes, int n_in,
                              void* d_out, int out_size, void* d_ws, size_t ws_size,
                              hipStream_t stream) {
  const float* x     = (const float*)d_in[0];
  const float* Wq    = (const float*)d_in[2];
  const float* bq    = (const float*)d_in[3];
  const float* Wk    = (const float*)d_in[4];
  const float* bk    = (const float*)d_in[5];
  const float* Wv    = (const float*)d_in[6];
  const float* bv    = (const float*)d_in[7];
  const float* Wd    = (const float*)d_in[8];
  const float* bd    = (const float*)d_in[9];
  const float* gamma = (const float*)d_in[10];
  const float* beta  = (const float*)d_in[11];
  float* out = (float*)d_out;

  char* ws = (char*)d_ws;
  const size_t SZ_T = (size_t)Mrows * Hq * 2;   // 64 MB bf16 tensor
  const size_t SZ_W = (size_t)Hq * Hq * 2;      // 2 MB bf16 weight
  bf16_t* xb   = (bf16_t*)(ws);
  bf16_t* qb   = (bf16_t*)(ws + SZ_T);
  bf16_t* kb   = (bf16_t*)(ws + 2 * SZ_T);
  bf16_t* vb   = (bf16_t*)(ws + 3 * SZ_T);
  bf16_t* ctxb = (bf16_t*)(ws + 4 * SZ_T);
  bf16_t* Wqb  = (bf16_t*)(ws + 5 * SZ_T);
  bf16_t* Wkb  = (bf16_t*)(ws + 5 * SZ_T + SZ_W);
  bf16_t* Wvb  = (bf16_t*)(ws + 5 * SZ_T + 2 * SZ_W);
  bf16_t* Wdb  = (bf16_t*)(ws + 5 * SZ_T + 3 * SZ_W);
  float*  ksum = (float*)(ws + 5 * SZ_T + 4 * SZ_W);                       // 32 KB
  float*  zbuf = (float*)(ws + 5 * SZ_T + 4 * SZ_W + (1 << 16));          // 2 MB
  bf16_t* kvb  = (bf16_t*)(ws + 5 * SZ_T + 4 * SZ_W + (1 << 16) + (size_t)Bq * Sq * NHq * 4);

  const int nX = Mrows * Hq;
  const int nW = Hq * Hq;

  lta_cvt_bf16<<<nX / 4 / 256, 256, 0, stream>>>(x,  xb,  nX);
  lta_cvt_bf16<<<nW / 4 / 256, 256, 0, stream>>>(Wq, Wqb, nW);
  lta_cvt_bf16<<<nW / 4 / 256, 256, 0, stream>>>(Wk, Wkb, nW);
  lta_cvt_bf16<<<nW / 4 / 256, 256, 0, stream>>>(Wv, Wvb, nW);
  lta_cvt_bf16<<<nW / 4 / 256, 256, 0, stream>>>(Wd, Wdb, nW);

  dim3 gGemm(Mrows / 128, Hq / 128);
  lta_gemm_proj<<<gGemm, 256, 0, stream>>>(xb, Wqb, bq, qb, 1);
  lta_gemm_proj<<<gGemm, 256, 0, stream>>>(xb, Wkb, bk, kb, 1);
  lta_gemm_proj<<<gGemm, 256, 0, stream>>>(xb, Wvb, bv, vb, 0);

  lta_ksum<<<Bq * NHq, 256, 0, stream>>>(kb, ksum);
  lta_z<<<(Bq * Sq * NHq) / 256, 256, 0, stream>>>(qb, ksum, zbuf);

  lta_kv<<<Bq * NHq, 128, 0, stream>>>(kb, vb, kvb);

  dim3 gCtx(Sq / 128, Bq * NHq);
  lta_ctx<<<gCtx, 256, 0, stream>>>(qb, kvb, zbuf, ctxb);

  lta_out<<<Mrows / 16, 256, 0, stream>>>(ctxb, Wdb, bd, x, gamma, beta, out);
}
